// MultiHeadAttention_10505490006017
// MI455X (gfx1250) — matrix operand order repaired and checked by reading, not compiled, not run on hardware
//
#include <hip/hip_runtime.h>
#include <hip/hip_bf16.h>
#include <cstdint>

typedef __attribute__((ext_vector_type(16))) __bf16 v16bf;
typedef __attribute__((ext_vector_type(8)))  float  v8f;
typedef __attribute__((ext_vector_type(4)))  unsigned int v4u;
typedef __attribute__((ext_vector_type(8)))  int    v8i;
typedef __attribute__((ext_vector_type(4)))  int    v4i;

__device__ __forceinline__ __bf16 f2bf(float f) {
  union { float f; uint32_t u; } c; c.f = f;
  uint32_t u = c.u + 0x7FFFu + ((c.u >> 16) & 1u);   // round-to-nearest-even
  union { unsigned short s; __bf16 b; } r; r.s = (unsigned short)(u >> 16);
  return r.b;
}

// ---------------------------------------------------------------------------
// TDM: issue a 2D tensor-tile load into LDS (D# groups per ISA 08 §8.3/8.4).
// dim0/dim1: tensor extents (elements), tile0/tile1: tile extents,
// stride0: row stride (elements). padI/padA: LDS padding encodings.
// ---------------------------------------------------------------------------
__device__ __forceinline__ void tdm_load_2d(
    uint32_t ldsAddr, const void* gptr,
    uint32_t dim0, uint32_t dim1, uint32_t tile0, uint32_t tile1,
    uint32_t stride0, uint32_t padI, uint32_t padA) {
  uint64_t ga = (uint64_t)(uintptr_t)gptr;
  v4u g0;
  g0.x = 1u;                                   // count = 1 (valid descriptor)
  g0.y = ldsAddr;                              // LDS byte address
  g0.z = (uint32_t)ga;                         // global addr [31:0]
  g0.w = (uint32_t)((ga >> 32) & 0x01FFFFFFu)  // global addr [56:32]
         | (2u << 30);                         // type = 2 ("image")
  v8i g1;
  g1[0] = (int)((1u << 16)                     // data_size = 1 (2 bytes)
              | (1u << 20)                     // pad_enable
              | (padI << 22) | (padA << 25));
  g1[1] = (int)((dim0 & 0xFFFFu) << 16);                         // tensor_dim0[15:0]
  g1[2] = (int)(((dim0 >> 16) & 0xFFFFu) | ((dim1 & 0xFFFFu) << 16));
  g1[3] = (int)(((dim1 >> 16) & 0xFFFFu) | (tile0 << 16));       // tile_dim0
  g1[4] = (int)tile1;                                            // tile_dim1 (tile_dim2=0)
  g1[5] = (int)stride0;                                          // tensor_dim0_stride[31:0]
  g1[6] = 0;                                                     // stride0 hi / stride1 lo
  g1[7] = 0;
  v4i gz = {0, 0, 0, 0};
#if __clang_major__ >= 23
  v8i gz8 = {0, 0, 0, 0, 0, 0, 0, 0};
  __builtin_amdgcn_tensor_load_to_lds(g0, g1, gz, gz, gz8, 0);
#else
  __builtin_amdgcn_tensor_load_to_lds(g0, g1, gz, gz, 0);
#endif
}

// ---------------------------------------------------------------------------
// Kernel: flat fp32 -> bf16 convert
// ---------------------------------------------------------------------------
__global__ void k_f32_to_bf16(const float* __restrict__ in,
                              __bf16* __restrict__ out, int n) {
  int i = blockIdx.x * blockDim.x + threadIdx.x;
  if (i < n) out[i] = f2bf(in[i]);
}

// ---------------------------------------------------------------------------
// Kernel: pack W[H, C, hd] fp32 -> bf16 [C, H*hd] (heads concatenated on cols)
// ---------------------------------------------------------------------------
__global__ void k_pack_w(const float* __restrict__ in, __bf16* __restrict__ out,
                         int C, int HD, int hd) {
  int i = blockIdx.x * blockDim.x + threadIdx.x;  // over C*HD
  if (i >= C * HD) return;
  int col = i % HD;
  int c   = i / HD;
  int h = col / hd, d = col % hd;
  out[(size_t)c * HD + col] = f2bf(in[(size_t)h * C * hd + (size_t)c * hd + d]);
}

// ---------------------------------------------------------------------------
// Tiled bf16 GEMM + bias: out[M,N] = A[M,K] * B[K,N] + bias[N]
// Block: 128 threads (4 waves). Block tile 64x64, wave tile 32x32, K-step 32.
// Tile staging via the Tensor Data Mover, double-buffered in LDS.
// LDS pitches (via TDM padding): A rows 16+4 DW = 80 B, B rows 32+4 DW = 144 B.
// ---------------------------------------------------------------------------
template <bool OUT_BF16>
__global__ __launch_bounds__(128) void k_gemm_bias(
    const __bf16* __restrict__ A, const __bf16* __restrict__ B,
    const float* __restrict__ bias, void* __restrict__ outp,
    int M, int N, int K) {
  constexpr int BM = 64, BN = 64, BK = 32;
  __shared__ __bf16 As[2][BM][BK + 8];   // pitch 40 elems = 80 B
  __shared__ __bf16 Bs[2][BK][BN + 8];   // pitch 72 elems = 144 B

  const int tid  = threadIdx.x;
  const int w    = tid >> 5;
  const int lane = tid & 31;
  const int l15  = lane & 15;
  const int halfsel = (lane < 16) ? 0 : 1;

  const int blockM = blockIdx.y * BM;
  const int blockN = blockIdx.x * BN;
  const int waveM = (w >> 1) * 32;
  const int waveN = (w & 1) * 32;

  const uint32_t asAddr[2] = {(uint32_t)(uintptr_t)&As[0][0][0],
                              (uint32_t)(uintptr_t)&As[1][0][0]};
  const uint32_t bsAddr[2] = {(uint32_t)(uintptr_t)&Bs[0][0][0],
                              (uint32_t)(uintptr_t)&Bs[1][0][0]};

  v8f c[2][2];
  c[0][0] = {}; c[0][1] = {}; c[1][0] = {}; c[1][1] = {};

  union AF { v16bf v; uint32_t u[8]; };
  union BF { v16bf v; uint32_t u[8]; __bf16 e[16]; };

  // Prologue: issue TDM loads for buffer 0 (A tile 64x32, B tile 32x64)
  if (tid == 0) {
    tdm_load_2d(asAddr[0], A + (size_t)blockM * K, (uint32_t)K, (uint32_t)M,
                BK, BM, (uint32_t)K, /*padI 16DW*/ 3, /*padA 4DW*/ 3);
    tdm_load_2d(bsAddr[0], B + blockN, (uint32_t)N, (uint32_t)K,
                BN, BK, (uint32_t)N, /*padI 32DW*/ 4, /*padA 4DW*/ 3);
  }

  int buf = 0;
  for (int k0 = 0; k0 < K; k0 += BK) {
    if (tid == 0) {
      if (k0 + BK < K) {  // issue next tile pair into the other buffer
        tdm_load_2d(asAddr[buf ^ 1], A + (size_t)blockM * K + (k0 + BK),
                    (uint32_t)K, (uint32_t)M, BK, BM, (uint32_t)K, 3, 3);
        tdm_load_2d(bsAddr[buf ^ 1], B + (size_t)(k0 + BK) * N + blockN,
                    (uint32_t)N, (uint32_t)K, BN, BK, (uint32_t)N, 4, 3);
        __builtin_amdgcn_s_wait_tensorcnt(2);  // current pair has landed
      } else {
        __builtin_amdgcn_s_wait_tensorcnt(0);
      }
    }
    __syncthreads();

    // Build fragments (contiguous k = 16*half + s in both A and B)
    AF a[2];
    const int kbA = halfsel ? 16 : 0;
#pragma unroll
    for (int mt = 0; mt < 2; ++mt) {
      int row = waveM + mt * 16 + l15;
#pragma unroll
      for (int j = 0; j < 4; ++j) {
        a[mt].u[j]     = *(const uint32_t*)&As[buf][row][kbA + 2 * j];
        a[mt].u[4 + j] = *(const uint32_t*)&As[buf][row][kbA + 8 + 2 * j];
      }
    }
    BF bfr[2];
    const int kbB = halfsel ? 16 : 0;
#pragma unroll
    for (int nt = 0; nt < 2; ++nt) {
      int col = waveN + nt * 16 + l15;
#pragma unroll
      for (int i = 0; i < 16; ++i) bfr[nt].e[i] = Bs[buf][kbB + i][col];
    }

#pragma unroll
    for (int mt = 0; mt < 2; ++mt)
#pragma unroll
      for (int nt = 0; nt < 2; ++nt)
        c[mt][nt] = __builtin_amdgcn_wmma_f32_16x16x32_bf16(
            false, a[mt].v, false, bfr[nt].v, (short)0, c[mt][nt], false, false);
    __syncthreads();
    buf ^= 1;
  }

  // Epilogue: C/D layout -> global (VGPR r: rows r / r+8)
  const int rowBase = halfsel * 8;
#pragma unroll
  for (int mt = 0; mt < 2; ++mt)
#pragma unroll
    for (int nt = 0; nt < 2; ++nt)
#pragma unroll
      for (int r = 0; r < 8; ++r) {
        int gr = blockM + waveM + mt * 16 + rowBase + r;
        int gc = blockN + waveN + nt * 16 + l15;
        float v = c[mt][nt][r] + bias[gc];
        if (OUT_BF16)
          ((__bf16*)outp)[(size_t)gr * N + gc] = f2bf(v);
        else
          ((float*)outp)[(size_t)gr * N + gc] = v;
      }
}

// ---------------------------------------------------------------------------
// Flash attention (causal, scale = C^-0.5 per the reference quirk).
// Block: 128 threads = 4 waves sharing (b,h); wave handles 16 queries.
// K/V tiles (32 keys x 64 hd) staged cooperatively into LDS per key block.
// Q/K/V/O are bf16 [B*T, H*hd] with col = h*hd + d.
// ---------------------------------------------------------------------------
__global__ __launch_bounds__(128) void k_flash_attn(
    const __bf16* __restrict__ Qm, const __bf16* __restrict__ Km,
    const __bf16* __restrict__ Vm, __bf16* __restrict__ Om,
    int Bn, int H, int T, int hd) {
  __shared__ __bf16 Kt[32][72];     // pitch 144 B
  __shared__ __bf16 Vt[32][72];
  __shared__ __bf16 Ps[4][16][40];  // per-wave P tile, pitch 80 B

  const int tid  = threadIdx.x;
  const int w    = tid >> 5;
  const int lane = tid & 31;
  const int l15  = lane & 15;
  const int halfsel = (lane < 16) ? 0 : 1;
  const int C = H * hd;

  const int nqb    = T / 64;
  const int bh     = blockIdx.x / nqb;
  const int qblock = (blockIdx.x % nqb) * 64;
  const int b = bh / H, h = bh % H;
  const int q0 = qblock + w * 16;

  const __bf16* Qbase = Qm + ((size_t)b * T) * C + h * hd;
  const __bf16* Kbase = Km + ((size_t)b * T) * C + h * hd;
  const __bf16* Vbase = Vm + ((size_t)b * T) * C + h * hd;

  union AF { v16bf v; uint32_t u[8]; };
  union BF { v16bf v; uint32_t u[8]; __bf16 e[16]; };

  // Preload Q A-fragments for the two hd K-steps (hd = 64 = 2 x 32)
  AF aq[2];
  {
    const __bf16* qrow = Qbase + (size_t)(q0 + l15) * C;
    const int kb = halfsel ? 16 : 0;
#pragma unroll
    for (int ks = 0; ks < 2; ++ks)
#pragma unroll
      for (int j = 0; j < 4; ++j) {
        aq[ks].u[j]     = *(const uint32_t*)(qrow + ks * 32 + kb + 2 * j);
        aq[ks].u[4 + j] = *(const uint32_t*)(qrow + ks * 32 + kb + 8 + 2 * j);
      }
  }

  v8f o[4];
  o[0] = {}; o[1] = {}; o[2] = {}; o[3] = {};
  float m[8], l[8];
#pragma unroll
  for (int r = 0; r < 8; ++r) { m[r] = -1e30f; l[r] = 0.0f; }

  const float scale = 0.03125f;  // 1/sqrt(C), C = 1024
  const int jmax = qblock + 63;  // uniform across the block (barrier-safe)

  for (int j0 = 0; j0 <= jmax; j0 += 32) {
    // ---- cooperative staging of K/V tiles (coalesced b128 loads) ----
#pragma unroll
    for (int li = 0; li < 2; ++li) {
      int idx = tid + li * 128;          // 0..255 -> 32 rows x 8 segs
      int row = idx >> 3, seg = idx & 7;
      *(uint4*)&Kt[row][seg * 8] =
          *(const uint4*)(Kbase + (size_t)(j0 + row) * C + seg * 8);
      *(uint4*)&Vt[row][seg * 8] =
          *(const uint4*)(Vbase + (size_t)(j0 + row) * C + seg * 8);
    }
    if (j0 + 32 <= jmax) {  // prefetch next key block
      __builtin_prefetch(Kbase + (size_t)(j0 + 32 + (tid >> 2)) * C, 0, 0);
      __builtin_prefetch(Vbase + (size_t)(j0 + 32 + (tid >> 2)) * C, 0, 0);
    }
    __syncthreads();

    // ---- S = Q * K^T for this 32-key block (two 16-key N-tiles) ----
    v8f s[2];
#pragma unroll
    for (int nt = 0; nt < 2; ++nt) {
      v8f acc = {};
      const int keyRow = nt * 16 + l15;
      const int kb = halfsel ? 16 : 0;
#pragma unroll
      for (int ks = 0; ks < 2; ++ks) {
        BF bk;
#pragma unroll
        for (int j = 0; j < 8; ++j)
          bk.u[j] = *(const uint32_t*)&Kt[keyRow][ks * 32 + kb + 2 * j];
        acc = __builtin_amdgcn_wmma_f32_16x16x32_bf16(
            false, aq[ks].v, false, bk.v, (short)0, acc, false, false);
      }
      s[nt] = acc;
    }

    // ---- online softmax (row = 16 lanes of a half-wave) ----
    float pv0[8], pv1[8];
#pragma unroll
    for (int r = 0; r < 8; ++r) {
      const int trow = q0 + halfsel * 8 + r;
      float s0 = s[0][r] * scale;
      float s1 = s[1][r] * scale;
      const bool msk0 = (j0 + l15) > trow;
      const bool msk1 = (j0 + 16 + l15) > trow;
      s0 = msk0 ? -1e30f : s0;
      s1 = msk1 ? -1e30f : s1;
      float mx = fmaxf(s0, s1);
      for (int off = 1; off < 16; off <<= 1)
        mx = fmaxf(mx, __shfl_xor(mx, off, 16));
      const float mnew = fmaxf(m[r], mx);
      const float sc = __expf(m[r] - mnew);
      const float p0 = msk0 ? 0.0f : __expf(s0 - mnew);
      const float p1 = msk1 ? 0.0f : __expf(s1 - mnew);
      float rs = p0 + p1;
      for (int off = 1; off < 16; off <<= 1) rs += __shfl_xor(rs, off, 16);
      l[r] = l[r] * sc + rs;
      m[r] = mnew;
#pragma unroll
      for (int nt2 = 0; nt2 < 4; ++nt2) o[nt2][r] *= sc;
      pv0[r] = p0; pv1[r] = p1;
    }

    // ---- P (C/D layout) -> LDS -> A-fragment layout ----
#pragma unroll
    for (int r = 0; r < 8; ++r) {
      const int Mr = halfsel * 8 + r;
      Ps[w][Mr][l15]      = f2bf(pv0[r]);
      Ps[w][Mr][16 + l15] = f2bf(pv1[r]);
    }
    __syncthreads();
    AF ap;
    {
      const int kb = halfsel ? 16 : 0;
#pragma unroll
      for (int j = 0; j < 4; ++j) {
        ap.u[j]     = *(const uint32_t*)&Ps[w][l15][kb + 2 * j];
        ap.u[4 + j] = *(const uint32_t*)&Ps[w][l15][kb + 8 + 2 * j];
      }
    }

    // ---- O += P * V  (K-dim = 32 keys, N = 64 hd -> 4 WMMAs) ----
    const int kb2 = halfsel ? 16 : 0;
#pragma unroll
    for (int nt2 = 0; nt2 < 4; ++nt2) {
      BF bv;
      const int n = nt2 * 16 + l15;
#pragma unroll
      for (int i = 0; i < 16; ++i) bv.e[i] = Vt[kb2 + i][n];
      o[nt2] = __builtin_amdgcn_wmma_f32_16x16x32_bf16(
          false, ap.v, false, bv.v, (short)0, o[nt2], false, false);
    }
    __syncthreads();  // protect Kt/Vt/Ps before next iteration's staging
  }

  // ---- epilogue: normalize and store bf16 ----
  __bf16* Obase = Om + ((size_t)b * T) * C + h * hd;
#pragma unroll
  for (int nt2 = 0; nt2 < 4; ++nt2)
#pragma unroll
    for (int r = 0; r < 8; ++r) {
      const int t = q0 + halfsel * 8 + r;
      const float inv = 1.0f / l[r];
      Obase[(size_t)t * C + nt2 * 16 + l15] = f2bf(o[nt2][r] * inv);
    }
}

// ---------------------------------------------------------------------------
// Launch
// ---------------------------------------------------------------------------
extern "C" void kernel_launch(void* const* d_in, const int* in_sizes, int n_in,
                              void* d_out, int out_size, void* d_ws, size_t ws_size,
                              hipStream_t stream) {
  const int Bn = 4, T = 2048, C = 1024, H = 16, hd = 64;
  const int M = Bn * T;  // 8192

  const float* x  = (const float*)d_in[0];
  const float* Wq = (const float*)d_in[1];
  const float* bq = (const float*)d_in[2];
  const float* Wk = (const float*)d_in[3];
  const float* bk = (const float*)d_in[4];
  const float* Wv = (const float*)d_in[5];
  const float* bv = (const float*)d_in[6];
  const float* Wp = (const float*)d_in[7];
  const float* bp = (const float*)d_in[8];

  char* ws = (char*)d_ws;
  size_t off = 0;
  __bf16* Xb  = (__bf16*)(ws + off); off += (size_t)M * C * 2;   // 16 MB
  __bf16* Wqb = (__bf16*)(ws + off); off += (size_t)C * C * 2;   //  2 MB
  __bf16* Wkb = (__bf16*)(ws + off); off += (size_t)C * C * 2;
  __bf16* Wvb = (__bf16*)(ws + off); off += (size_t)C * C * 2;
  __bf16* Wpb = (__bf16*)(ws + off); off += (size_t)C * C * 2;
  __bf16* Qb  = (__bf16*)(ws + off); off += (size_t)M * C * 2;   // 16 MB
  __bf16* Kb  = (__bf16*)(ws + off); off += (size_t)M * C * 2;
  __bf16* Vb  = (__bf16*)(ws + off); off += (size_t)M * C * 2;
  __bf16* Ob  = (__bf16*)(ws + off); off += (size_t)M * C * 2;   // total ~88 MB

  const int nx = M * C;
  k_f32_to_bf16<<<(nx + 255) / 256, 256, 0, stream>>>(x, Xb, nx);
  const int nw = C * C;
  k_pack_w<<<(nw + 255) / 256, 256, 0, stream>>>(Wq, Wqb, C, C, hd);
  k_pack_w<<<(nw + 255) / 256, 256, 0, stream>>>(Wk, Wkb, C, C, hd);
  k_pack_w<<<(nw + 255) / 256, 256, 0, stream>>>(Wv, Wvb, C, C, hd);
  k_f32_to_bf16<<<(nw + 255) / 256, 256, 0, stream>>>(Wp, Wpb, nw);

  dim3 gg(C / 64, M / 64);  // 16 x 128
  k_gemm_bias<true><<<gg, 128, 0, stream>>>(Xb, Wqb, bq, (void*)Qb, M, C, C);
  k_gemm_bias<true><<<gg, 128, 0, stream>>>(Xb, Wkb, bk, (void*)Kb, M, C, C);
  k_gemm_bias<true><<<gg, 128, 0, stream>>>(Xb, Wvb, bv, (void*)Vb, M, C, C);

  k_flash_attn<<<Bn * H * (T / 64), 128, 0, stream>>>(Qb, Kb, Vb, Ob, Bn, H, T, hd);

  k_gemm_bias<false><<<gg, 128, 0, stream>>>(Ob, Wpb, bp, d_out, M, C, C);
}
